// GeToEdgeAggregator_55473797595471
// MI455X (gfx1250) — matrix operand order repaired and checked by reading, not compiled, not run on hardware
//
#include <hip/hip_runtime.h>
#include <hip/hip_bf16.h>
#include <math.h>

typedef __attribute__((ext_vector_type(16))) __bf16 bf16x16;
typedef __attribute__((ext_vector_type(8)))  float  f32x8;

namespace {
constexpr int kB    = 8192;
constexpr int kN    = 20;
constexpr int kDin  = 256;
constexpr int kGin  = 128;
constexpr int kH    = 603;
constexpr int kHP   = 608;   // H padded to multiple of 16
constexpr int kDout = 256;

// ---- workspace layout (bytes) ----
constexpr size_t WS_WT_FEAT   = 0;                                     // [608][256] bf16
constexpr size_t WS_WT_GETO   = WS_WT_FEAT   + (size_t)kHP * kDin * 2; // [608][128] bf16
constexpr size_t WS_WT_CONCAT = WS_WT_GETO   + (size_t)kHP * kGin * 2; // [608][128] bf16
constexpr size_t WS_WT_CG     = WS_WT_CONCAT + (size_t)kHP * kGin * 2; // [512][608] bf16
constexpr size_t WS_WT_NF     = WS_WT_CG     + (size_t)2 * kDout * kHP * 2; // [256][608]
constexpr size_t WS_WT_NG     = WS_WT_NF     + (size_t)kDout * kHP * 2;     // [256][608]
constexpr size_t WS_WT_SF     = WS_WT_NG     + (size_t)kDout * kHP * 2;     // [256][256]
constexpr size_t WS_WT_SG     = WS_WT_SF     + (size_t)kDout * kDin * 2;    // [256][128]
constexpr size_t WS_FEAT_POOL = WS_WT_SG     + (size_t)kDout * kGin * 2;    // [B][608] f32
constexpr size_t WS_GETO_POOL = WS_FEAT_POOL + (size_t)kB * kHP * 4;        // [B][608] f32
constexpr size_t WS_CONC_POOL = WS_GETO_POOL + (size_t)kB * kHP * 4;        // [B][608] f32
} // namespace

// ---------------- WMMA helpers (gfx1250 wave32 layouts) ----------------

__device__ __forceinline__ f32x8 wmma_bf16(bf16x16 a, bf16x16 b, f32x8 c) {
  return __builtin_amdgcn_wmma_f32_16x16x32_bf16(false, a, false, b, (short)0, c,
                                                 false, false);
}

// A fragment (16x32 bf16, M x K). Lane l: row M = l&15, holds K runs
// [kk*32 + 8*l2, +8) and [kk*32 + 16 + 8*l2, +8).  Two b128 loads.
__device__ __forceinline__ bf16x16 load_a_frag(const __bf16* arow, int kk, int l2) {
  union { bf16x16 v; uint4 q[2]; } u;
  const __bf16* p = arow + kk * 32 + 8 * l2;
  u.q[0] = *reinterpret_cast<const uint4*>(p);
  u.q[1] = *reinterpret_cast<const uint4*>(p + 16);
  return u.v;
}

// B fragment (32x16 bf16, K x N) read from a TRANSPOSED weight row
// (row = output column h, contiguous in K). Lane l: col N = l&15, holds
// K runs [kk*32 + 8*l2, +8) and [kk*32 + 16 + 8*l2, +8).  Two b128 loads.
__device__ __forceinline__ bf16x16 load_b_frag(const __bf16* brow, int kk, int l2) {
  union { bf16x16 v; uint4 q[2]; } u;
  const __bf16* p = brow + kk * 32 + 8 * l2;
  u.q[0] = *reinterpret_cast<const uint4*>(p);
  u.q[1] = *reinterpret_cast<const uint4*>(p + 16);
  return u.v;
}

// C/D layout: element r of f32x8 -> row M = r + 8*l2, col N = lane&15.

template <int K>
__device__ __forceinline__ f32x8 tile_from_lds(const bf16x16* af, const __bf16* wt,
                                               int t, int m, int l2) {
  const __bf16* brow = wt + (size_t)(t * 16 + m) * K;
  f32x8 c = {};
#pragma unroll
  for (int kk = 0; kk < K / 32; ++kk)
    c = wmma_bf16(af[kk], load_b_frag(brow, kk, l2), c);
  return c;
}

template <int KT>
__device__ __forceinline__ f32x8 gemm_gl(const __bf16* arow, const __bf16* bglob,
                                         int ldb, int tile, int m, int l2) {
  const __bf16* brow = bglob + (size_t)(tile * 16 + m) * ldb;
  f32x8 c = {};
#pragma unroll
  for (int kk = 0; kk < KT; ++kk)
    c = wmma_bf16(load_a_frag(arow, kk, l2), load_b_frag(brow, kk, l2), c);
  return c;
}

// ---------------- weight prep: f32 [R][C] -> bf16 transposed [Cp][Rp] ----------------

__global__ __launch_bounds__(256) void transpose_bf16_kernel(
    const float* __restrict__ src, __bf16* __restrict__ dst,
    int R, int C, int Rp, int Cp) {
  const int idx = blockIdx.x * 256 + threadIdx.x;
  if (idx >= Rp * Cp) return;
  const int c = idx / Rp;
  const int r = idx - c * Rp;
  float v = (r < R && c < C) ? src[(size_t)r * C + c] : 0.0f;
  dst[idx] = (__bf16)v;
}

// ---------------- stage A: relu( A[b,n,:] @ W + bias ), mean over n ----------------
// One workgroup = 16 batch rows, full H (608 cols).  Whole transposed weight
// lives in LDS (304KB for K=256 -> uses the 320KB CDNA5 WGP LDS).

template <int K, int NN>
__global__ __launch_bounds__(512, 1) void pool_relu_kernel(
    const float* __restrict__ A, const float* __restrict__ bias,
    const __bf16* __restrict__ WT, float* __restrict__ outp) {
  extern __shared__ char smem[];
  __bf16* wt_lds = reinterpret_cast<__bf16*>(smem);                        // [608][K]
  __bf16* a_lds  = reinterpret_cast<__bf16*>(smem + (size_t)kHP * K * 2);  // [16][K]

  const int tid = threadIdx.x;
  const int lane = tid & 31, wid = tid >> 5;
  const int l2 = lane >> 4, m = lane & 15;
  const int b0 = blockIdx.x * 16;

  {  // whole transposed weight -> LDS, 128-bit copies
    const uint4* src = reinterpret_cast<const uint4*>(WT);
    uint4* dst = reinterpret_cast<uint4*>(wt_lds);
    constexpr int nq = kHP * K / 8;
    for (int i = tid; i < nq; i += 512) dst[i] = src[i];
  }

  const int t0 = wid, t1 = wid + 16, t2 = wid + 32;
  const bool has2 = (t2 < kHP / 16);  // wave-uniform
  float bv0, bv1, bv2 = 0.0f;
  { int h = t0 * 16 + m; bv0 = (h < kH) ? bias[h] : 0.0f; }
  { int h = t1 * 16 + m; bv1 = (h < kH) ? bias[h] : 0.0f; }
  if (has2) { int h = t2 * 16 + m; bv2 = (h < kH) ? bias[h] : 0.0f; }

  f32x8 p0 = {}, p1 = {}, p2 = {};
  __syncthreads();

  for (int n = 0; n < NN; ++n) {
    for (int i = tid; i < 16 * K; i += 512) {  // stage 16 x K A-tile as bf16
      const int r = i / K, k = i - r * K;
      a_lds[i] = (__bf16)A[((size_t)(b0 + r) * NN + n) * K + k];
    }
    __syncthreads();

    bf16x16 af[K / 32];
    const __bf16* arow = a_lds + (size_t)m * K;
#pragma unroll
    for (int kk = 0; kk < K / 32; ++kk) af[kk] = load_a_frag(arow, kk, l2);

    f32x8 c = tile_from_lds<K>(af, wt_lds, t0, m, l2);
#pragma unroll
    for (int r = 0; r < 8; ++r) p0[r] += fmaxf(c[r] + bv0, 0.0f);
    c = tile_from_lds<K>(af, wt_lds, t1, m, l2);
#pragma unroll
    for (int r = 0; r < 8; ++r) p1[r] += fmaxf(c[r] + bv1, 0.0f);
    if (has2) {
      c = tile_from_lds<K>(af, wt_lds, t2, m, l2);
#pragma unroll
      for (int r = 0; r < 8; ++r) p2[r] += fmaxf(c[r] + bv2, 0.0f);
    }
    __syncthreads();
  }

  const float sc = 1.0f / (float)NN;
#pragma unroll
  for (int r = 0; r < 8; ++r) {
    const size_t row = (size_t)(b0 + r + 8 * l2);
    outp[row * kHP + t0 * 16 + m] = p0[r] * sc;
    outp[row * kHP + t1 * 16 + m] = p1[r] * sc;
    if (has2) outp[row * kHP + t2 * 16 + m] = p2[r] * sc;
  }
}

// ---------------- stage A': softmax branch ----------------
// 4 batches per workgroup: 84 live rows (20 neigh + self each) -> 6 M-tiles.
// Logits kept in LDS as bf16 (needed whole-row before normalization).

__global__ __launch_bounds__(512, 1) void pool_softmax_kernel(
    const float* __restrict__ self_geto, const float* __restrict__ neigh_geto,
    const float* __restrict__ bias, const __bf16* __restrict__ WT,
    float* __restrict__ outp) {
  extern __shared__ char smem[];
  __bf16* wt_lds = reinterpret_cast<__bf16*>(smem);                              // [608][128]
  __bf16* a_lds  = reinterpret_cast<__bf16*>(smem + (size_t)kHP * kGin * 2);     // [96][128]
  __bf16* lg     = reinterpret_cast<__bf16*>(smem + (size_t)(kHP + 96) * kGin * 2); // [96][608]
  float*  scale  = reinterpret_cast<float*>(smem + (size_t)(kHP + 96) * kGin * 2 +
                                            (size_t)96 * kHP * 2);               // [96]

  const int tid = threadIdx.x;
  const int lane = tid & 31, wid = tid >> 5;
  const int l2 = lane >> 4, m = lane & 15;
  const int b0 = blockIdx.x * 4;

  {
    const uint4* src = reinterpret_cast<const uint4*>(WT);
    uint4* dst = reinterpret_cast<uint4*>(wt_lds);
    constexpr int nq = kHP * kGin / 8;
    for (int i = tid; i < nq; i += 512) dst[i] = src[i];
  }
  for (int i = tid; i < 96 * kGin; i += 512) {  // rows: 21 per batch; 84..95 zero
    const int r = i >> 7, k = i & 127;
    float v = 0.0f;
    if (r < 84) {
      const int bl = r / 21, rr = r - bl * 21;
      v = (rr < kN) ? neigh_geto[((size_t)(b0 + bl) * kN + rr) * kGin + k]
                    : self_geto[(size_t)(b0 + bl) * kGin + k];
    }
    a_lds[i] = (__bf16)v;
  }
  __syncthreads();

  for (int nt = wid; nt < kHP / 16; nt += 16) {  // wave-uniform
    const __bf16* brow = wt_lds + (size_t)(nt * 16 + m) * kGin;
    bf16x16 bf[4];
#pragma unroll
    for (int kk = 0; kk < 4; ++kk) bf[kk] = load_b_frag(brow, kk, l2);
    const int h = nt * 16 + m;
    const float bv = (h < kH) ? bias[h] : 0.0f;
#pragma unroll
    for (int mt = 0; mt < 6; ++mt) {
      const __bf16* arow = a_lds + (size_t)(mt * 16 + m) * kGin;
      f32x8 c = {};
#pragma unroll
      for (int kk = 0; kk < 4; ++kk)
        c = wmma_bf16(load_a_frag(arow, kk, l2), bf[kk], c);
#pragma unroll
      for (int r = 0; r < 8; ++r)
        lg[(size_t)(mt * 16 + r + 8 * l2) * kHP + h] = (__bf16)(c[r] + bv);
    }
  }
  __syncthreads();

  // per-row softmax (wave handles whole rows; shfl_xor reductions, wave32)
  for (int j = wid; j < 84; j += 16) {
    __bf16* row = lg + (size_t)j * kHP;
    float mx = -INFINITY;
    for (int h = lane; h < kH; h += 32) mx = fmaxf(mx, (float)row[h]);
#pragma unroll
    for (int off = 16; off > 0; off >>= 1) mx = fmaxf(mx, __shfl_xor(mx, off, 32));
    float sm = 0.0f;
    for (int h = lane; h < kHP; h += 32) {
      float e = (h < kH) ? __expf((float)row[h] - mx) : 0.0f;
      row[h] = (__bf16)e;
      sm += e;
    }
#pragma unroll
    for (int off = 16; off > 0; off >>= 1) sm += __shfl_xor(sm, off, 32);
    if (lane == 0) scale[j] = 1.0f / (sm * 21.0f);  // fold mean over 21 rows
  }
  __syncthreads();

  for (int i = tid; i < 4 * kHP; i += 512) {
    const int bl = i / kHP, h = i - bl * kHP;
    float acc = 0.0f;
#pragma unroll
    for (int r = 0; r < 21; ++r) {
      const int j = bl * 21 + r;
      acc += (float)lg[(size_t)j * kHP + h] * scale[j];
    }
    outp[(size_t)(b0 + bl) * kHP + h] = acc;
  }
}

// ---------------- stage B: projections + fused elementwise epilogues ----------------

__global__ __launch_bounds__(256, 1) void output_kernel(
    const float* __restrict__ gp, const float* __restrict__ fp,
    const float* __restrict__ gp2, const float* __restrict__ self_vecs,
    const float* __restrict__ self_geto,
    const __bf16* __restrict__ WT_cg, const __bf16* __restrict__ WT_nf,
    const __bf16* __restrict__ WT_sf, const __bf16* __restrict__ WT_ng,
    const __bf16* __restrict__ WT_sg,
    float* __restrict__ node_out, float* __restrict__ geto_out) {
  extern __shared__ char smem[];
  __bf16* gp_l  = reinterpret_cast<__bf16*>(smem);  // [16][608]
  __bf16* fp_l  = gp_l + 16 * kHP;
  __bf16* gp2_l = fp_l + 16 * kHP;
  __bf16* sv_l  = gp2_l + 16 * kHP;                 // [16][256]
  __bf16* sg_l  = sv_l + 16 * kDin;                 // [16][128]

  const int tid = threadIdx.x;
  const int lane = tid & 31, wid = tid >> 5;
  const int l2 = lane >> 4, m = lane & 15;
  const int b0 = blockIdx.x * 16;

  for (int i = tid; i < 16 * kHP; i += 256) {
    const int r = i / kHP, k = i - r * kHP;
    const size_t off = (size_t)(b0 + r) * kHP + k;
    gp_l[i]  = (__bf16)gp[off];
    fp_l[i]  = (__bf16)fp[off];
    gp2_l[i] = (__bf16)gp2[off];
  }
  for (int i = tid; i < 16 * kDin; i += 256) {
    const int r = i >> 8, k = i & 255;
    sv_l[i] = (__bf16)self_vecs[(size_t)(b0 + r) * kDin + k];
  }
  for (int i = tid; i < 16 * kGin; i += 256) {
    const int r = i >> 7, k = i & 127;
    sg_l[i] = (__bf16)self_geto[(size_t)(b0 + r) * kGin + k];
  }
  __syncthreads();

  for (int t = wid; t < 48; t += 8) {  // 32 node col-tiles + 16 geto col-tiles
    if (t < 32) {
      f32x8 e = gemm_gl<kHP / 32>(gp_l + (size_t)m * kHP, WT_cg, kHP, t, m, l2);
      f32x8 o;
      if (t < 16)
        o = gemm_gl<kDin / 32>(sv_l + (size_t)m * kDin, WT_sf, kDin, t, m, l2);
      else
        o = gemm_gl<kHP / 32>(fp_l + (size_t)m * kHP, WT_nf, kHP, t - 16, m, l2);
#pragma unroll
      for (int r = 0; r < 8; ++r) {
        const float v = e[r] * o[r];
        node_out[(size_t)(b0 + r + 8 * l2) * (2 * kDout) + t * 16 + m] =
            v > 0.0f ? v : 0.0f;
      }
    } else {
      const int t2 = t - 32;
      f32x8 g = gemm_gl<kHP / 32>(gp2_l + (size_t)m * kHP, WT_ng, kHP, t2, m, l2);
      f32x8 s = gemm_gl<kGin / 32>(sg_l + (size_t)m * kGin, WT_sg, kGin, t2, m, l2);
#pragma unroll
      for (int r = 0; r < 8; ++r) {
        const float v = g[r] + s[r];
        geto_out[(size_t)(b0 + r + 8 * l2) * kDout + t2 * 16 + m] =
            v > 0.0f ? v : 0.0f;
      }
    }
  }
}

// ---------------- host launcher ----------------

extern "C" void kernel_launch(void* const* d_in, const int* in_sizes, int n_in,
                              void* d_out, int out_size, void* d_ws, size_t ws_size,
                              hipStream_t stream) {
  (void)in_sizes; (void)n_in; (void)out_size; (void)ws_size;

  const float* self_vecs  = (const float*)d_in[0];
  const float* neigh_vecs = (const float*)d_in[1];
  const float* self_geto  = (const float*)d_in[2];
  const float* neigh_geto = (const float*)d_in[3];
  const float* W_feat     = (const float*)d_in[4];
  const float* b_feat     = (const float*)d_in[5];
  const float* W_geto     = (const float*)d_in[6];
  const float* b_geto     = (const float*)d_in[7];
  const float* W_concat   = (const float*)d_in[8];
  const float* b_concat   = (const float*)d_in[9];
  const float* self_feat_w   = (const float*)d_in[10];
  const float* self_geto_w   = (const float*)d_in[11];
  const float* combined_geto = (const float*)d_in[12];
  const float* neigh_feat_w  = (const float*)d_in[13];
  const float* neigh_geto_w  = (const float*)d_in[14];

  char* ws = (char*)d_ws;
  __bf16* wt_feat   = (__bf16*)(ws + WS_WT_FEAT);
  __bf16* wt_geto   = (__bf16*)(ws + WS_WT_GETO);
  __bf16* wt_concat = (__bf16*)(ws + WS_WT_CONCAT);
  __bf16* wt_cg     = (__bf16*)(ws + WS_WT_CG);
  __bf16* wt_nf     = (__bf16*)(ws + WS_WT_NF);
  __bf16* wt_ng     = (__bf16*)(ws + WS_WT_NG);
  __bf16* wt_sf     = (__bf16*)(ws + WS_WT_SF);
  __bf16* wt_sg     = (__bf16*)(ws + WS_WT_SG);
  float* feat_pool = (float*)(ws + WS_FEAT_POOL);
  float* geto_pool = (float*)(ws + WS_GETO_POOL);
  float* conc_pool = (float*)(ws + WS_CONC_POOL);

  float* node_out = (float*)d_out;                        // [B, 512]
  float* geto_out = node_out + (size_t)kB * 2 * kDout;    // [B, 256]

  auto tlaunch = [&](const float* src, __bf16* dst, int R, int C, int Rp, int Cp) {
    const int tot = Rp * Cp;
    transpose_bf16_kernel<<<(tot + 255) / 256, 256, 0, stream>>>(src, dst, R, C, Rp, Cp);
  };
  tlaunch(W_feat,        wt_feat,   kDin, kH,    kDin, kHP);   // [608][256]
  tlaunch(W_geto,        wt_geto,   kGin, kH,    kGin, kHP);   // [608][128]
  tlaunch(W_concat,      wt_concat, kGin, kH,    kGin, kHP);   // [608][128]
  tlaunch(combined_geto, wt_cg,     kH, 2 * kDout, kHP, 2 * kDout); // [512][608]
  tlaunch(neigh_feat_w,  wt_nf,     kH, kDout,   kHP, kDout);  // [256][608]
  tlaunch(neigh_geto_w,  wt_ng,     kH, kDout,   kHP, kDout);  // [256][608]
  tlaunch(self_feat_w,   wt_sf,     kDin, kDout, kDin, kDout); // [256][256]
  tlaunch(self_geto_w,   wt_sg,     kGin, kDout, kGin, kDout); // [256][128]

  const size_t smem_feat = (size_t)kHP * kDin * 2 + (size_t)16 * kDin * 2; // 319488
  pool_relu_kernel<kDin, kN><<<kB / 16, 512, smem_feat, stream>>>(
      neigh_vecs, b_feat, wt_feat, feat_pool);

  const size_t smem_geto = (size_t)kHP * kGin * 2 + (size_t)16 * kGin * 2; // 159744
  pool_relu_kernel<kGin, kN><<<kB / 16, 512, smem_geto, stream>>>(
      neigh_geto, b_geto, wt_geto, geto_pool);

  const size_t smem_sm = (size_t)(kHP + 96) * kGin * 2 + (size_t)96 * kHP * 2 +
                         96 * sizeof(float);                              // 297344
  pool_softmax_kernel<<<kB / 4, 512, smem_sm, stream>>>(
      self_geto, neigh_geto, b_concat, wt_concat, conc_pool);

  const size_t smem_out = (size_t)16 * (3 * kHP + kDin + kGin) * 2;       // 70656
  output_kernel<<<kB / 16, 256, smem_out, stream>>>(
      conc_pool, feat_pool, geto_pool, self_vecs, self_geto,
      wt_cg, wt_nf, wt_sf, wt_ng, wt_sg, node_out, geto_out);
}
